// GatedLinearAttention_38517266710658
// MI455X (gfx1250) — matrix operand order repaired and checked by reading, not compiled, not run on hardware
//
#include <hip/hip_runtime.h>
#include <hip/hip_bf16.h>

// ---------------- problem constants ----------------
constexpr int B_  = 2;
constexpr int L_  = 4096;
constexpr int D_  = 2048;
constexpr int H_  = 16;
constexpr int DK_ = 64;
constexpr int DF_ = 64;
constexpr int DV_ = 128;
constexpr int P_  = 2 * DF_;   // 128
constexpr int C_  = 64;        // chunk
constexpr int BL_ = B_ * L_;   // 8192
constexpr float EPS_      = 1e-12f;
constexpr float GATE_NORM_= 16.0f;
constexpr float RMS_EPS_  = 1e-5f;

// ---------------- CDNA5 feature detection ----------------
#if defined(__has_builtin)
#if __has_builtin(__builtin_amdgcn_tensor_load_to_lds)
#define HAVE_TDM 1
#endif
#endif
#ifndef HAVE_TDM
#define HAVE_TDM 0
#endif

// ---------------- WMMA plumbing ----------------
typedef __attribute__((ext_vector_type(16))) __bf16 bfrag;
typedef __attribute__((ext_vector_type(8)))  float  v8f;
typedef __attribute__((ext_vector_type(4)))  unsigned int v4u;
typedef __attribute__((ext_vector_type(8)))  int v8i;
typedef __attribute__((ext_vector_type(4)))  int v4i;

__device__ __forceinline__ v8f wmma_bf16(bfrag a, bfrag b, v8f c) {
  return __builtin_amdgcn_wmma_f32_16x16x32_bf16(false, a, false, b, (short)0, c, false, false);
}

// A fragment: 16x32 (MxK), element (m,k) at p[m*sm + k*sk], scaled.
// ISA layout: lanes 0-15: m=lane, V0-3 -> k=0..7, V4-7 -> k=16..23 (pairs);
//             lanes16-31: m=lane-16, k offset +8 in each group.
template <typename T>
__device__ __forceinline__ bfrag load_frag_a(const T* p, int sm, int sk, float scale) {
  const int lane = threadIdx.x & 31;
  const T* q = p + (lane & 15) * sm + ((lane >> 4) << 3) * sk;
  bfrag f;
#pragma unroll
  for (int v = 0; v < 8; ++v) {
    const int k = ((v & 3) << 1) + ((v >> 2) << 4);
    f[2 * v]     = (__bf16)((float)q[k * sk] * scale);
    f[2 * v + 1] = (__bf16)((float)q[(k + 1) * sk] * scale);
  }
  return f;
}

// B fragment: 32x16 (KxN), element (k,n) at p[k*sk + n*sn].
// ISA layout: lanes 0-15: n=lane, V0-3 -> k=0..7, V4-7 -> k=16..23 (pairs);
//             lanes16-31: k offset +8 in each group (matches A).
template <typename T>
__device__ __forceinline__ bfrag load_frag_b(const T* p, int sk, int sn) {
  const int lane = threadIdx.x & 31;
  const T* q = p + (lane & 15) * sn + ((lane >> 4) << 3) * sk;
  bfrag f;
#pragma unroll
  for (int v = 0; v < 8; ++v) {
    const int k = ((v & 3) << 1) + ((v >> 2) << 4);
    f[2 * v]     = (__bf16)(float)q[k * sk];
    f[2 * v + 1] = (__bf16)(float)q[(k + 1) * sk];
  }
  return f;
}

// ---------------- Tensor Data Mover helpers ----------------
__device__ __forceinline__ unsigned lds_off(const void* p) {
  // generic LDS pointer: low 32 bits are the LDS byte offset
  return (unsigned)(unsigned long long)p;
}

#if HAVE_TDM
// DMA a fully in-bounds 2-D bf16 tile (tile_d0 fast-dim elements x tile_d1 rows,
// row stride = stride_elems) from global memory to LDS, packed row-major.
// tensor dims are set equal to tile dims so the OOB logic never triggers.
__device__ __forceinline__ void tdm_load_tile_bf16(unsigned lds_addr, const __bf16* gsrc,
                                                   unsigned tile_d0, unsigned tile_d1,
                                                   unsigned long long stride_elems) {
  const unsigned long long ga = (unsigned long long)gsrc;
  v4u g0;
  g0[0] = 1u;                                            // count=1 (valid), user mode
  g0[1] = lds_addr;                                      // D#.lds_addr
  g0[2] = (unsigned)ga;                                  // global_addr[31:0]
  g0[3] = (unsigned)((ga >> 32) & 0x01ffffffu) | (2u << 30); // addr[56:32] | type=2
  v8i g1;
  g1[0] = 1 << 16;                                       // data_size=1 (2 bytes), mask=0
  g1[1] = (int)((tile_d0 & 0xffffu) << 16);              // tensor_dim0[15:0] @ bits 63:48
  g1[2] = (int)(((tile_d0 >> 16) & 0xffffu) |            // tensor_dim0[31:16]
                ((tile_d1 & 0xffffu) << 16));            // tensor_dim1[15:0]
  g1[3] = (int)(((tile_d1 >> 16) & 0xffffu) |            // tensor_dim1[31:16]
                ((tile_d0 & 0xffffu) << 16));            // tile_dim0
  g1[4] = (int)(tile_d1 & 0xffffu);                      // tile_dim1 (tile_dim2=0)
  g1[5] = (int)(unsigned)(stride_elems & 0xffffffffull); // tensor_dim0_stride[31:0]
  g1[6] = (int)(unsigned)((stride_elems >> 32) & 0xffffull); // stride[47:32], dim1_stride=0
  g1[7] = 0;
  v4i gz = {0, 0, 0, 0};
#if __clang_major__ >= 23
  v8i gz8 = {0, 0, 0, 0, 0, 0, 0, 0};
  __builtin_amdgcn_tensor_load_to_lds(g0, g1, gz, gz, gz8, 0);
#else
  __builtin_amdgcn_tensor_load_to_lds(g0, g1, gz, gz, 0);
#endif
}
#endif

// ---------------- cast kernel ----------------
__global__ __launch_bounds__(256) void k_cast_bf16(const float* __restrict__ in,
                                                   __bf16* __restrict__ out, long n) {
  long i = (long)blockIdx.x * blockDim.x + threadIdx.x;
  const long stride = (long)gridDim.x * blockDim.x;
  for (; i < n; i += stride) out[i] = (__bf16)in[i];
}

// ---------------- generic tiled bf16 WMMA GEMM (TDM double-buffered) ----------------
// C[M,N] = A[M,K] * B[K,N]; A,B row-major bf16; C f32 or bf16.
// grid.z batches with element strides (strideA/B/C).
template <int BM, int BN, int WM, int WN, bool OUT_BF16>
__global__ __launch_bounds__(256) void k_gemm_bf16(
    const __bf16* __restrict__ A, const __bf16* __restrict__ Bm, void* __restrict__ Cm,
    int M, int N, int K, int lda, int ldb, int ldc,
    long strideA, long strideB, long strideC) {
  constexpr int BK = 32;
  __shared__ __align__(16) __bf16 As[2][BM * BK];
  __shared__ __align__(16) __bf16 Bs[2][BK * BN];
  const int tid  = threadIdx.x;
  const int wave = tid >> 5;
  const int lane = tid & 31;
  A  += (long)blockIdx.z * strideA;
  Bm += (long)blockIdx.z * strideB;
  const int bm = blockIdx.y * BM;
  const int bn = blockIdx.x * BN;
  constexpr int WAVES_N = BN / WN;
  const int wm = (wave / WAVES_N) * WM;
  const int wn = (wave % WAVES_N) * WN;
  constexpr int FM = WM / 16, FN = WN / 16;

  v8f acc[FM][FN];
#pragma unroll
  for (int fm = 0; fm < FM; ++fm)
#pragma unroll
    for (int fn = 0; fn < FN; ++fn)
#pragma unroll
      for (int e = 0; e < 8; ++e) acc[fm][fn][e] = 0.0f;

  const int T = K / BK;
#if HAVE_TDM
  const bool issuer = (wave == 0);   // wave-uniform: one TDM issue per tile
  if (issuer) {
    tdm_load_tile_bf16(lds_off(&As[0][0]), A + (long)bm * lda, BK, BM, (unsigned long long)lda);
    tdm_load_tile_bf16(lds_off(&Bs[0][0]), Bm + bn, BN, BK, (unsigned long long)ldb);
    __builtin_amdgcn_s_wait_tensorcnt(0);
  }
  __syncthreads();
#endif

  for (int i = 0; i < T; ++i) {
#if HAVE_TDM
    const int buf = i & 1;
    if (issuer && (i + 1) < T) {      // prefetch next tile into the other buffer
      const int k1 = (i + 1) * BK;
      tdm_load_tile_bf16(lds_off(&As[buf ^ 1][0]), A + (long)bm * lda + k1,
                         BK, BM, (unsigned long long)lda);
      tdm_load_tile_bf16(lds_off(&Bs[buf ^ 1][0]), Bm + (long)k1 * ldb + bn,
                         BN, BK, (unsigned long long)ldb);
    }
#else
    const int buf = 0;
    const int k0 = i * BK;
    for (int ii = tid; ii < BM * BK / 8; ii += 256) {
      const int r = ii / (BK / 8);
      const int c = (ii % (BK / 8)) * 8;
      *(uint4*)&As[0][r * BK + c] = *(const uint4*)&A[(long)(bm + r) * lda + k0 + c];
    }
    for (int ii = tid; ii < BK * BN / 8; ii += 256) {
      const int r = ii / (BN / 8);
      const int c = (ii % (BN / 8)) * 8;
      *(uint4*)&Bs[0][r * BN + c] = *(const uint4*)&Bm[(long)(k0 + r) * ldb + bn + c];
    }
    __syncthreads();
#endif
    bfrag afr[FM], bfr[FN];
#pragma unroll
    for (int fm = 0; fm < FM; ++fm)
      afr[fm] = load_frag_a(&As[buf][(wm + fm * 16) * BK], BK, 1, 1.0f);
#pragma unroll
    for (int fn = 0; fn < FN; ++fn)
      bfr[fn] = load_frag_b(&Bs[buf][wn + fn * 16], BN, 1);
#pragma unroll
    for (int fm = 0; fm < FM; ++fm)
#pragma unroll
      for (int fn = 0; fn < FN; ++fn)
        acc[fm][fn] = wmma_bf16(afr[fm], bfr[fn], acc[fm][fn]);
#if HAVE_TDM
    if (issuer && (i + 1) < T) __builtin_amdgcn_s_wait_tensorcnt(0);
    __syncthreads();
#else
    __syncthreads();
#endif
  }

  // epilogue: C frag layout -> lanes 0-15: m=r, n=lane; lanes 16-31: m=8+r
  const int cn  = lane & 15;
  const int rhi = (lane >> 4) * 8;
#pragma unroll
  for (int fm = 0; fm < FM; ++fm)
#pragma unroll
    for (int fn = 0; fn < FN; ++fn)
#pragma unroll
      for (int r = 0; r < 8; ++r) {
        const long row = bm + wm + fm * 16 + rhi + r;
        const long col = bn + wn + fn * 16 + cn;
        const long idx = (long)blockIdx.z * strideC + row * ldc + col;
        if (OUT_BF16) ((__bf16*)Cm)[idx] = (__bf16)acc[fm][fn][r];
        else          ((float*)Cm)[idx]  = acc[fm][fn][r];
      }
  (void)M; (void)N;
}

// ---------------- RoPE (q,k f32 -> rotated bf16) ----------------
__global__ __launch_bounds__(256) void k_rope(const float* __restrict__ q,
                                              const float* __restrict__ k,
                                              __bf16* __restrict__ qo,
                                              __bf16* __restrict__ ko,
                                              const int* __restrict__ pos) {
  const long total = (long)BL_ * (H_ * DK_);
  long i = (long)blockIdx.x * blockDim.x + threadIdx.x;
  const long stride = (long)gridDim.x * blockDim.x;
  for (; i < total; i += stride) {
    const long row = i >> 10;               // H*DK = 1024
    const int  d   = (int)(i & 63);         // within head dim
    const int  l   = (int)(row & (L_ - 1)); // L power of 2
    const int  fidx = d & 31;
    const float invf = __powf(10000.0f, -(float)fidx * (1.0f / 32.0f));
    const float ang  = (float)pos[l] * invf;
    float s, c;
    __sincosf(ang, &s, &c);
    const float qv = q[i], kv = k[i];
    float qr, kr;
    if (d < 32) { qr = qv * c - q[i + 32] * s; kr = kv * c - k[i + 32] * s; }
    else        { qr = qv * c + q[i - 32] * s; kr = kv * c + k[i - 32] * s; }
    qo[i] = (__bf16)qr;
    ko[i] = (__bf16)kr;
  }
}

// ---------------- feature map: softmax(z) ++ softmax(-z), clip, scale ----------------
// z: (BL, H*DF) f32 -> out: (B,H,L,P) bf16. One wave per (row, head).
__global__ __launch_bounds__(256) void k_featuremap(const float* __restrict__ z,
                                                    __bf16* __restrict__ outp, float scale) {
  const int gw   = blockIdx.x * (blockDim.x >> 5) + (threadIdx.x >> 5);
  const int lane = threadIdx.x & 31;
  if (gw >= BL_ * H_) return;
  const int  h   = gw & (H_ - 1);
  const long row = gw >> 4;
  const long b   = row >> 12;        // row / L
  const long l   = row & (L_ - 1);
  const float* zp = z + row * (H_ * DF_) + h * DF_;
  const float z0 = zp[2 * lane], z1 = zp[2 * lane + 1];
  float mx = fmaxf(z0, z1), mn = fminf(z0, z1);
#pragma unroll
  for (int off = 16; off > 0; off >>= 1) {
    mx = fmaxf(mx, __shfl_xor(mx, off));
    mn = fminf(mn, __shfl_xor(mn, off));
  }
  const float e0 = expf(z0 - mx), e1 = expf(z1 - mx);   // softmax(z)
  const float f0 = expf(mn - z0), f1 = expf(mn - z1);   // softmax(-z)
  float es = e0 + e1, fs = f0 + f1;
#pragma unroll
  for (int off = 16; off > 0; off >>= 1) {
    es += __shfl_xor(es, off);
    fs += __shfl_xor(fs, off);
  }
  __bf16* op = outp + ((b * H_ + h) * L_ + l) * P_;
  op[2 * lane]          = (__bf16)(fmaxf(e0 / es, EPS_) * scale);
  op[2 * lane + 1]      = (__bf16)(fmaxf(e1 / es, EPS_) * scale);
  op[64 + 2 * lane]     = (__bf16)(fmaxf(f0 / fs, EPS_) * scale);
  op[64 + 2 * lane + 1] = (__bf16)(fmaxf(f1 / fs, EPS_) * scale);
}

// ---------------- gate MLP: g = log_sigmoid(x@Wg1@Wg2 + bg2)/16 -> (B,H,L,P) ----------------
__global__ __launch_bounds__(256) void k_gate(const float* __restrict__ x,
                                              const float* __restrict__ Wg1,
                                              const float* __restrict__ Wg2,
                                              const float* __restrict__ bg2,
                                              float* __restrict__ g) {
  __shared__ float part[256];
  __shared__ float hv[16];
  const int row = blockIdx.x;     // b*L + l
  const int tid = threadIdx.x;
  const float* xr = x + (long)row * D_;
  const int j    = tid & 15;
  const int gidx = tid >> 4;
  float acc = 0.0f;
  for (int i = gidx; i < D_; i += 16) acc += xr[i] * Wg1[i * 16 + j];
  part[tid] = acc;
  __syncthreads();
  if (tid < 16) {
    float s = 0.0f;
#pragma unroll
    for (int gg = 0; gg < 16; ++gg) s += part[gg * 16 + tid];
    hv[tid] = s;
  }
  __syncthreads();
  const int b = row / L_, l = row % L_;
  for (int c = tid; c < H_ * P_; c += 256) {
    float a = bg2[c];
#pragma unroll
    for (int jj = 0; jj < 16; ++jj) a += hv[jj] * Wg2[jj * (H_ * P_) + c];
    const float ls = fminf(a, 0.0f) - log1pf(expf(-fabsf(a)));   // log_sigmoid
    const int h = c >> 7, p = c & 127;
    g[(((long)b * H_ + h) * L_ + l) * P_ + p] = ls * (1.0f / GATE_NORM_);
  }
}

// ---------------- GLA chunkwise scan ----------------
// One workgroup per (b,h); S (P x DV) lives in LDS across the 64-chunk scan.
__global__ __launch_bounds__(256) void k_gla(const __bf16* __restrict__ qfeat,
                                             const __bf16* __restrict__ kfeat,
                                             const __bf16* __restrict__ vb,
                                             const float* __restrict__ g,
                                             float* __restrict__ o) {
  __shared__ __align__(16) float  S[P_ * DV_];    // 64KB state
  __shared__ __align__(16) float  bC[C_ * P_];    // 32KB cumsum(g)
  __shared__ __align__(16) __bf16 qbS[C_ * P_];   // 16KB q*exp(b)
  __shared__ __align__(16) __bf16 kbS[C_ * P_];   // 16KB k*exp(-b)
  __shared__ __align__(16) __bf16 vS[C_ * DV_];   // 16KB
  __shared__ __align__(16) float  aS[C_ * C_];    // 16KB masked A

  const int bh   = blockIdx.x;            // b*H + h
  const int tid  = threadIdx.x;
  const int lane = tid & 31;
  const int wave = tid >> 5;
  const int bq = bh / H_, h = bh % H_;
  const long base = (long)bh * L_;

  for (int i = tid; i < P_ * DV_; i += 256) S[i] = 0.0f;
  __syncthreads();

  for (int ch = 0; ch < L_ / C_; ++ch) {
    const long r0 = base + (long)ch * C_;

    // prefetch next chunk's streams while this chunk computes
    if (ch + 1 < L_ / C_) {
      const long rn = r0 + C_;
      for (int t = tid; t < C_; t += 256) {
        __builtin_prefetch(&g[(rn + t) * P_], 0, 1);
        __builtin_prefetch(&qfeat[(rn + t) * P_], 0, 1);
        __builtin_prefetch(&kfeat[(rn + t) * P_], 0, 1);
      }
    }

    // 1. cumsum of g along t (one column per thread)
    if (tid < P_) {
      float a = 0.0f;
      for (int t = 0; t < C_; ++t) { a += g[(r0 + t) * P_ + tid]; bC[t * P_ + tid] = a; }
    }
    __syncthreads();

    // 2. qb = q*exp(b), kb = k*exp(-b), stage v
    for (int i = tid; i < C_ * P_; i += 256) {
      const int t = i >> 7, p = i & 127;
      const float bb = bC[t * P_ + p];
      qbS[i] = (__bf16)((float)qfeat[(r0 + t) * P_ + p] * expf(bb));
      kbS[i] = (__bf16)((float)kfeat[(r0 + t) * P_ + p] * expf(-bb));
      vS[i]  = vb[((long)bq * L_ + ch * C_ + t) * (H_ * DV_) + h * DV_ + p];
    }
    __syncthreads();

    // 3. A = qb @ kb^T (64x64, K=128), causal mask. 16 tiles -> 2 per wave.
#pragma unroll
    for (int it = 0; it < 2; ++it) {
      const int tIdx = wave * 2 + it;
      const int tm = (tIdx >> 2) * 16, tn = (tIdx & 3) * 16;
      v8f acc;
#pragma unroll
      for (int e = 0; e < 8; ++e) acc[e] = 0.0f;
#pragma unroll
      for (int kk = 0; kk < P_; kk += 32) {
        bfrag a   = load_frag_a(&qbS[tm * P_ + kk], P_, 1, 1.0f);
        bfrag bfr = load_frag_b(&kbS[tn * P_ + kk], 1, P_);   // kb^T
        acc = wmma_bf16(a, bfr, acc);
      }
      const int cn = lane & 15, rhi = (lane >> 4) * 8;
#pragma unroll
      for (int r = 0; r < 8; ++r) {
        const int m = tm + rhi + r, n = tn + cn;
        aS[m * C_ + n] = (m >= n) ? acc[r] : 0.0f;
      }
    }
    __syncthreads();

    // 4. o = A @ v + qb @ S  (64x128). 32 tiles -> 4 per wave.
#pragma unroll
    for (int it = 0; it < 4; ++it) {
      const int tIdx = wave * 4 + it;
      const int tm = (tIdx >> 3) * 16, tn = (tIdx & 7) * 16;
      v8f acc;
#pragma unroll
      for (int e = 0; e < 8; ++e) acc[e] = 0.0f;
#pragma unroll
      for (int kk = 0; kk < C_; kk += 32) {          // intra-chunk
        bfrag a   = load_frag_a(&aS[tm * C_ + kk], C_, 1, 1.0f);
        bfrag bfr = load_frag_b(&vS[kk * DV_ + tn], DV_, 1);
        acc = wmma_bf16(a, bfr, acc);
      }
#pragma unroll
      for (int kk = 0; kk < P_; kk += 32) {          // inter-chunk via state
        bfrag a   = load_frag_a(&qbS[tm * P_ + kk], P_, 1, 1.0f);
        bfrag bfr = load_frag_b(&S[kk * DV_ + tn], DV_, 1);
        acc = wmma_bf16(a, bfr, acc);
      }
      const int cn = lane & 15, rhi = (lane >> 4) * 8;
#pragma unroll
      for (int r = 0; r < 8; ++r) {
        const long l = (long)ch * C_ + tm + rhi + r;
        o[((long)bq * L_ + l) * (H_ * DV_) + h * DV_ + tn + cn] = acc[r];
      }
    }
    __syncthreads();

    // 5. S = diag(exp(blast)) * S + (kb*exp(blast))^T @ v. 64 tiles -> 8 per wave.
#pragma unroll
    for (int it = 0; it < 8; ++it) {
      const int tIdx = wave * 8 + it;
      const int tm = (tIdx >> 3) * 16, tn = (tIdx & 7) * 16;
      const int cn = lane & 15, rhi = (lane >> 4) * 8;
      v8f acc;
#pragma unroll
      for (int r = 0; r < 8; ++r) {
        const int m = tm + rhi + r;
        acc[r] = S[m * DV_ + tn + cn] * expf(bC[(C_ - 1) * P_ + m]);
      }
      const float sc = expf(bC[(C_ - 1) * P_ + tm + (lane & 15)]); // per-lane row p
#pragma unroll
      for (int kk = 0; kk < C_; kk += 32) {
        bfrag a   = load_frag_a(&kbS[kk * P_ + tm], 1, P_, sc);   // kd^T, row-scaled
        bfrag bfr = load_frag_b(&vS[kk * DV_ + tn], DV_, 1);
        acc = wmma_bf16(a, bfr, acc);
      }
#pragma unroll
      for (int r = 0; r < 8; ++r) S[(tm + rhi + r) * DV_ + tn + cn] = acc[r];
    }
    __syncthreads();
  }
}

// ---------------- RMS norm * gnorm_w * silu(gate) -> bf16 ----------------
__global__ __launch_bounds__(256) void k_rmsgate(const float* __restrict__ o,
                                                 const float* __restrict__ gatep,
                                                 const float* __restrict__ gnorm_w,
                                                 __bf16* __restrict__ yb) {
  const int gw   = blockIdx.x * (blockDim.x >> 5) + (threadIdx.x >> 5);
  const int lane = threadIdx.x & 31;
  if (gw >= BL_ * H_) return;
  const long rowh = gw;                 // (b*L+l)*H + h
  const int  h    = (int)(rowh & (H_ - 1));
  const float* op = o + rowh * DV_;
  float vals[4];
  float ss = 0.0f;
#pragma unroll
  for (int i = 0; i < 4; ++i) {
    vals[i] = op[lane + 32 * i];
    ss += vals[i] * vals[i];
  }
#pragma unroll
  for (int off = 16; off > 0; off >>= 1) ss += __shfl_xor(ss, off);
  const float rn = rsqrtf(ss * (1.0f / DV_) + RMS_EPS_);
#pragma unroll
  for (int i = 0; i < 4; ++i) {
    const int dv = lane + 32 * i;
    const float gt = gatep[rowh * DV_ + dv];
    const float y  = vals[i] * rn * gnorm_w[h * DV_ + dv] * gt * (1.0f / (1.0f + expf(-gt)));
    yb[rowh * DV_ + dv] = (__bf16)y;
  }
}

// ---------------- host ----------------
extern "C" void kernel_launch(void* const* d_in, const int* in_sizes, int n_in,
                              void* d_out, int out_size, void* d_ws, size_t ws_size,
                              hipStream_t stream) {
  const float* x      = (const float*)d_in[0];
  const float* Wq     = (const float*)d_in[1];
  const float* Wk     = (const float*)d_in[2];
  const float* Wv     = (const float*)d_in[3];
  const float* Wphi_q = (const float*)d_in[4];
  const float* Wphi_k = (const float*)d_in[5];
  const float* Wg1    = (const float*)d_in[6];
  const float* Wg2    = (const float*)d_in[7];
  const float* bg2    = (const float*)d_in[8];
  const float* Wog    = (const float*)d_in[9];
  const float* gnorm  = (const float*)d_in[10];
  const float* Wout   = (const float*)d_in[11];
  const int*   pos    = (const int*)d_in[12];
  float* out = (float*)d_out;

  char* ws = (char*)d_ws;
  size_t off = 0;
  auto take = [&](size_t bytes) -> char* {
    off = (off + 255) & ~(size_t)255;
    char* p = ws + off;
    off += bytes;
    return p;
  };

  __bf16* xb    = (__bf16*)take((size_t)BL_ * D_ * 2);
  __bf16* Wqb   = (__bf16*)take((size_t)D_ * H_ * DK_ * 2);
  __bf16* Wkb   = (__bf16*)take((size_t)D_ * H_ * DK_ * 2);
  __bf16* Wvb   = (__bf16*)take((size_t)D_ * H_ * DV_ * 2);
  __bf16* Wogb  = (__bf16*)take((size_t)D_ * H_ * DV_ * 2);
  __bf16* Woutb = (__bf16*)take((size_t)H_ * DV_ * D_ * 2);
  __bf16* Wpqb  = (__bf16*)take((size_t)H_ * DK_ * DF_ * 2);
  __bf16* Wpkb  = (__bf16*)take((size_t)H_ * DK_ * DF_ * 2);
  float*  qproj = (float*)take((size_t)BL_ * H_ * DK_ * 4);   // reused as zq
  float*  kproj = (float*)take((size_t)BL_ * H_ * DK_ * 4);   // reused as zk
  __bf16* vb    = (__bf16*)take((size_t)BL_ * H_ * DV_ * 2);
  float*  gatep = (float*)take((size_t)BL_ * H_ * DV_ * 4);
  __bf16* qrope = (__bf16*)take((size_t)BL_ * H_ * DK_ * 2);
  __bf16* krope = (__bf16*)take((size_t)BL_ * H_ * DK_ * 2);
  __bf16* qfeat = (__bf16*)take((size_t)B_ * H_ * L_ * P_ * 2);
  __bf16* kfeat = (__bf16*)take((size_t)B_ * H_ * L_ * P_ * 2);
  float*  gbuf  = (float*)take((size_t)B_ * H_ * L_ * P_ * 4);
  float*  obuf  = (float*)take((size_t)BL_ * H_ * DV_ * 4);
  float*  zq = qproj;
  float*  zk = kproj;
  __bf16* yb = xb;   // xb dead after projection GEMMs

  auto cast = [&](const float* src, __bf16* dst, long n) {
    k_cast_bf16<<<dim3(2048), 256, 0, stream>>>(src, dst, n);
  };
  cast(x,      xb,    (long)BL_ * D_);
  cast(Wq,     Wqb,   (long)D_ * H_ * DK_);
  cast(Wk,     Wkb,   (long)D_ * H_ * DK_);
  cast(Wv,     Wvb,   (long)D_ * H_ * DV_);
  cast(Wog,    Wogb,  (long)D_ * H_ * DV_);
  cast(Wout,   Woutb, (long)H_ * DV_ * D_);
  cast(Wphi_q, Wpqb,  (long)H_ * DK_ * DF_);
  cast(Wphi_k, Wpkb,  (long)H_ * DK_ * DF_);

  // projections
  k_gemm_bf16<128,128,32,64,false><<<dim3((H_*DK_)/128, BL_/128, 1), 256, 0, stream>>>(
      xb, Wqb, qproj, BL_, H_*DK_, D_, D_, H_*DK_, H_*DK_, 0, 0, 0);
  k_gemm_bf16<128,128,32,64,false><<<dim3((H_*DK_)/128, BL_/128, 1), 256, 0, stream>>>(
      xb, Wkb, kproj, BL_, H_*DK_, D_, D_, H_*DK_, H_*DK_, 0, 0, 0);
  k_gemm_bf16<128,128,32,64,true><<<dim3((H_*DV_)/128, BL_/128, 1), 256, 0, stream>>>(
      xb, Wvb, vb, BL_, H_*DV_, D_, D_, H_*DV_, H_*DV_, 0, 0, 0);
  k_gemm_bf16<128,128,32,64,false><<<dim3((H_*DV_)/128, BL_/128, 1), 256, 0, stream>>>(
      xb, Wogb, gatep, BL_, H_*DV_, D_, D_, H_*DV_, H_*DV_, 0, 0, 0);

  // RoPE
  k_rope<<<dim3(16384), 256, 0, stream>>>(qproj, kproj, qrope, krope, pos);

  // per-head phi GEMMs (batched over heads via grid.z)
  k_gemm_bf16<64,64,32,16,false><<<dim3(1, BL_/64, H_), 256, 0, stream>>>(
      qrope, Wpqb, zq, BL_, DF_, DK_, H_*DK_, DF_, H_*DF_,
      (long)DK_, (long)DK_*DF_, (long)DF_);
  k_gemm_bf16<64,64,32,16,false><<<dim3(1, BL_/64, H_), 256, 0, stream>>>(
      krope, Wpkb, zk, BL_, DF_, DK_, H_*DK_, DF_, H_*DF_,
      (long)DK_, (long)DK_*DF_, (long)DF_);

  // feature maps (q gets P^-0.5)
  k_featuremap<<<dim3(BL_ * H_ / 8), 256, 0, stream>>>(zq, qfeat, 0.08838834764831845f);
  k_featuremap<<<dim3(BL_ * H_ / 8), 256, 0, stream>>>(zk, kfeat, 1.0f);

  // gate MLP
  k_gate<<<dim3(BL_), 256, 0, stream>>>(x, Wg1, Wg2, bg2, gbuf);

  // GLA chunkwise scan
  k_gla<<<dim3(B_ * H_), 256, 0, stream>>>(qfeat, kfeat, vb, gbuf, obuf);

  // RMS norm + silu gate
  k_rmsgate<<<dim3(BL_ * H_ / 8), 256, 0, stream>>>(obuf, gatep, gnorm, yb);

  // final projection into d_out
  k_gemm_bf16<128,128,32,64,false><<<dim3(D_/128, BL_/128, 1), 256, 0, stream>>>(
      yb, Woutb, out, BL_, D_, H_*DV_, H_*DV_, D_, D_, 0, 0, 0);

  (void)in_sizes; (void)n_in; (void)out_size; (void)ws_size;
}
